// NSATransformerEncoder_82858509074964
// MI455X (gfx1250) — matrix operand order repaired and checked by reading, not compiled, not run on hardware
//
#include <hip/hip_runtime.h>
#include <stdint.h>

// ---------------------------------------------------------------------------
// CDNA5 (gfx1250) NSA transformer encoder forward.
// All large GEMMs go through v_wmma_f32_16x16x32_bf16 (wave32 WMMA),
// 2x2 register-blocked, with weights pre-transposed for contiguous B packs.
// ---------------------------------------------------------------------------

typedef __attribute__((ext_vector_type(16))) __bf16 v16bf;
typedef __attribute__((ext_vector_type(8)))  float  v8f;

namespace cfg {
constexpr int L  = 2,  B = 2, S = 2048, D = 512, H = 8, DH = 64, FF = 2048;
constexpr int CW = 4,  CS = 2, SW = 4,  NSEL = 2, WIN = 2;
constexpr int N   = B * S;                // 4096 token rows
constexpr int NC  = (S - CW) / CS + 1;    // 1023 compression chunks
constexpr int NCP = NC + 1;               // 1024 slots (mem slot + chunks)
constexpr int NR  = B * H * NC;           // 16368 compression GEMM rows
constexpr int G3H = 3 * H;                // 24 gate columns
}

__device__ __constant__ float kNEG = -1e30f;

// Native RNE f32->bf16 (lowers to v_cvt_pk_bf16_f32 when pairs fuse).
static __device__ __forceinline__ __bf16 f2bf(float f) { return (__bf16)f; }

// A-matrix (16x32 bf16) per-lane pack. Lane = 16*half + m.
// VGPR j in 0..3 holds K = 8*half + 2j (+1); j in 4..7 holds K = 16 + 8*half + 2(j-4).
static __device__ __forceinline__ v16bf pack_a(const float* Ar, int hf) {
  v16bf a;
#pragma unroll
  for (int j = 0; j < 8; ++j) {
    int base = (j < 4) ? (8 * hf + 2 * j) : (16 + 8 * hf + 2 * (j - 4));
    a[2 * j]     = f2bf(Ar[base]);
    a[2 * j + 1] = f2bf(Ar[base + 1]);
  }
  return a;
}

// B-matrix (32x16 bf16) pack from an N-major source (row = column n, contiguous K).
// Lane column n = lane%16; element i holds K = 8*half + i (i<8), 16 + 8*half + (i-8) (i>=8).
static __device__ __forceinline__ v16bf pack_bN(const float* Br, int hf) {
  v16bf b;
#pragma unroll
  for (int i = 0; i < 16; ++i) b[i] = f2bf(Br[(i < 8) ? (8 * hf + i) : (16 + 8 * hf + i - 8)]);
  return b;
}

// ---------------------------------------------------------------------------
// Blocked bf16 WMMA GEMM: C[N,M] = epi(A[N,K] @ W[K,M]), W given transposed
// as WT[M,K]. Each wave computes (16*NT) x (16*MT) outputs -> NT*MT WMMAs per
// 32-deep K step amortize NT+MT packs.
// EPI: 0 = none, 1 = +bias, 2 = +bias then exact GELU, 3 = +bias + residual.
// ---------------------------------------------------------------------------
template <int EPI, int NT, int MT>
__global__ void __launch_bounds__(32)
gemm_bf16_wmma(const float* __restrict__ A, const float* __restrict__ WT,
               const float* __restrict__ bias, const float* __restrict__ res,
               float* __restrict__ C, int K, int M) {
  const int lane = threadIdx.x;
  const int mrow = lane & 15, hf = lane >> 4;
  const int c0 = blockIdx.x * 16 * MT;
  const int r0 = blockIdx.y * 16 * NT;

  v8f acc[NT][MT];
#pragma unroll
  for (int i = 0; i < NT; ++i)
#pragma unroll
    for (int j = 0; j < MT; ++j) acc[i][j] = (v8f){};

  for (int k0 = 0; k0 < K; k0 += 32) {
    v16bf a[NT], b[MT];
#pragma unroll
    for (int i = 0; i < NT; ++i)
      a[i] = pack_a(A + (size_t)(r0 + 16 * i + mrow) * K + k0, hf);
#pragma unroll
    for (int j = 0; j < MT; ++j)
      b[j] = pack_bN(WT + (size_t)(c0 + 16 * j + mrow) * K + k0, hf);
#pragma unroll
    for (int i = 0; i < NT; ++i)
#pragma unroll
      for (int j = 0; j < MT; ++j)
        acc[i][j] = __builtin_amdgcn_wmma_f32_16x16x32_bf16(
            false, a[i], false, b[j], (short)0, acc[i][j], false, false);
  }

#pragma unroll
  for (int i = 0; i < NT; ++i) {
#pragma unroll
    for (int j = 0; j < MT; ++j) {
#pragma unroll
      for (int r = 0; r < 8; ++r) {
        int row = r0 + 16 * i + r + 8 * hf;
        int col = c0 + 16 * j + mrow;
        float v = acc[i][j][r];
        if (EPI >= 1) v += bias[col];
        if (EPI == 2) v = 0.5f * v * (1.0f + erff(v * 0.70710678118654752440f));
        if (EPI == 3) v += res[(size_t)row * M + col];
        C[(size_t)row * M + col] = v;
      }
    }
  }
}

// Weight transpose W[K,M] -> WT[M,K] (contiguous reads, done once per GEMM).
__global__ void __launch_bounds__(256)
transpose_w(const float* __restrict__ W, float* __restrict__ WT, int K, int M) {
  size_t idx = (size_t)blockIdx.x * blockDim.x + threadIdx.x;
  if (idx >= (size_t)K * M) return;
  int k = (int)(idx / M), m = (int)(idx % M);
  WT[(size_t)m * K + k] = W[idx];
}

// ---------------------------------------------------------------------------
// Norm kernels: one 256-thread block per row of D=512.
// ---------------------------------------------------------------------------
__global__ void __launch_bounds__(256)
layernorm_kernel(const float* __restrict__ x, const float* __restrict__ g,
                 const float* __restrict__ b, float* __restrict__ out) {
  __shared__ float red[256];
  const size_t row = blockIdx.x;
  const int t = threadIdx.x;
  const float* xr = x + row * cfg::D;
  float a0 = xr[t], a1 = xr[t + 256];
  red[t] = a0 + a1;
  __syncthreads();
  for (int o = 128; o > 0; o >>= 1) { if (t < o) red[t] += red[t + o]; __syncthreads(); }
  float mu = red[0] * (1.0f / cfg::D);
  __syncthreads();
  float d0 = a0 - mu, d1 = a1 - mu;
  red[t] = d0 * d0 + d1 * d1;
  __syncthreads();
  for (int o = 128; o > 0; o >>= 1) { if (t < o) red[t] += red[t + o]; __syncthreads(); }
  float rstd = rsqrtf(red[0] * (1.0f / cfg::D) + 1e-5f);
  out[row * cfg::D + t]       = d0 * rstd * g[t]       + b[t];
  out[row * cfg::D + t + 256] = d1 * rstd * g[t + 256] + b[t + 256];
}

__global__ void __launch_bounds__(256)
rmsnorm_kernel(const float* __restrict__ x, const float* __restrict__ g,
               float* __restrict__ out) {
  __shared__ float red[256];
  const size_t row = blockIdx.x;
  const int t = threadIdx.x;
  const float* xr = x + row * cfg::D;
  float a0 = xr[t], a1 = xr[t + 256];
  red[t] = a0 * a0 + a1 * a1;
  __syncthreads();
  for (int o = 128; o > 0; o >>= 1) { if (t < o) red[t] += red[t + o]; __syncthreads(); }
  float rms = rsqrtf(red[0] * (1.0f / cfg::D) + 1e-6f);
  out[row * cfg::D + t]       = a0 * rms * g[t];
  out[row * cfg::D + t + 256] = a1 * rms * g[t + 256];
}

// ---------------------------------------------------------------------------
// RoPE on q and k (per-head, half = DH/2 = 32).
// ---------------------------------------------------------------------------
__global__ void __launch_bounds__(256)
rope_kernel(const float* __restrict__ q, const float* __restrict__ k,
            float* __restrict__ qr, float* __restrict__ kr) {
  size_t idx = (size_t)blockIdx.x * blockDim.x + threadIdx.x;
  if (idx >= (size_t)cfg::B * cfg::S * cfg::H * 32) return;
  int i = idx & 31;
  size_t rem = idx >> 5;
  int h = rem % cfg::H;
  size_t bs = rem / cfg::H;                 // b*S + s
  int s = (int)(bs % cfg::S);
  float fr  = powf(10000.0f, -(float)i * (1.0f / 32.0f));
  float ang = (float)s * fr;
  float c = cosf(ang), sn = sinf(ang);
  size_t base = bs * cfg::D + (size_t)h * cfg::DH;
  float q1 = q[base + i], q2 = q[base + i + 32];
  qr[base + i]      = q1 * c - q2 * sn;
  qr[base + i + 32] = q1 * sn + q2 * c;
  float k1 = k[base + i], k2 = k[base + i + 32];
  kr[base + i]      = k1 * c - k2 * sn;
  kr[base + i + 32] = k1 * sn + k2 * c;
}

// ---------------------------------------------------------------------------
// Build compression rows kb/vb: (B,H,NC, CW*DH) = k[b, 2c+w, h, dh] + k_pos[h,w,dh]
// ---------------------------------------------------------------------------
__global__ void __launch_bounds__(256)
build_comp_rows(const float* __restrict__ k, const float* __restrict__ v,
                const float* __restrict__ k_pos, const float* __restrict__ v_pos,
                float* __restrict__ kbr, float* __restrict__ vbr) {
  size_t idx = (size_t)blockIdx.x * blockDim.x + threadIdx.x;
  if (idx >= (size_t)cfg::B * cfg::H * cfg::NC * (cfg::CW * cfg::DH)) return;
  int j = idx & 255;                    // CW*DH = 256
  size_t rem = idx >> 8;
  int c = rem % cfg::NC;
  size_t bh = rem / cfg::NC;
  int h = bh % cfg::H;
  int b = (int)(bh / cfg::H);
  int w = j >> 6, dh = j & 63;
  int tok = c * cfg::CS + w;
  size_t src = ((size_t)b * cfg::S + tok) * cfg::D + (size_t)h * cfg::DH + dh;
  size_t pos = ((size_t)h * cfg::CW + w) * cfg::DH + dh;
  kbr[idx] = k[src] + k_pos[pos];
  vbr[idx] = v[src] + v_pos[pos];
}

// Prepend mem slot. ck keeps (B,H,NCP,DH); cv is emitted TRANSPOSED as
// cvT (B,H,DH,NCP) so the P @ CV WMMA streams its B operand contiguously.
__global__ void __launch_bounds__(256)
assemble_comp(const float* __restrict__ ckc, const float* __restrict__ cvc,
              const float* __restrict__ mem_k, const float* __restrict__ mem_v,
              float* __restrict__ ck, float* __restrict__ cvT) {
  size_t idx = (size_t)blockIdx.x * blockDim.x + threadIdx.x;
  if (idx >= (size_t)cfg::B * cfg::H * cfg::NCP * cfg::DH) return;
  int dh = idx & 63;
  size_t rem = idx >> 6;
  int n = rem % cfg::NCP;
  size_t bh = rem / cfg::NCP;
  int h = bh % cfg::H;
  float kv, vv;
  if (n == 0) {
    kv = mem_k[(size_t)h * cfg::DH + dh];
    vv = mem_v[(size_t)h * cfg::DH + dh];
  } else {
    size_t srow = bh * cfg::NC + (n - 1);
    kv = ckc[srow * cfg::DH + dh];
    vv = cvc[srow * cfg::DH + dh];
  }
  ck[idx] = kv;
  cvT[(bh * cfg::DH + dh) * cfg::NCP + n] = vv;
}

// ---------------------------------------------------------------------------
// Compressed attention: one wave per (b, h, 16-query tile).
//  phase 1: WMMA sim(16x1024) = Q(16x64) @ CK^T into LDS (A tiles hoisted)
//  phase 2: masked softmax per row + analytic top-2 block selection
//           (ov is banded: sel[m] = imp[2m] + 0.5*imp[2m-1] + 0.5*imp[2m+1])
//  phase 3: WMMA c_out(16x64) = P(16x1024) @ CV(1024x64), CV pre-transposed
// ---------------------------------------------------------------------------
__global__ void __launch_bounds__(32)
cattn_kernel(const float* __restrict__ q, const float* __restrict__ ck,
             const float* __restrict__ cvT, float* __restrict__ c_out,
             int* __restrict__ topi) {
  extern __shared__ float P[];                 // 16 x 1024 f32 = 64 KB
  constexpr int ST = cfg::S / 16;              // 128 query tiles per (b,h)
  const int wid = blockIdx.x;
  const int b  = wid / (cfg::H * ST);
  const int hh = (wid / ST) % cfg::H;
  const int s0 = (wid % ST) * 16;
  const int lane = threadIdx.x;
  const int mrow = lane & 15, hf = lane >> 4;
  const float scale = 0.125f;                  // DH^-0.5

  const float* qb   = q   + (size_t)b * cfg::S * cfg::D + (size_t)hh * cfg::DH;
  const float* ckb  = ck  + ((size_t)b * cfg::H + hh) * cfg::NCP * cfg::DH;
  const float* cvTb = cvT + ((size_t)b * cfg::H + hh) * cfg::DH * cfg::NCP;

  // phase 1
  const float* Ar = qb + (size_t)(s0 + mrow) * cfg::D;
  const v16bf a0 = pack_a(Ar, hf);             // K = 0..31 (loop invariant)
  const v16bf a1 = pack_a(Ar + 32, hf);        // K = 32..63
  for (int c0 = 0; c0 < cfg::NCP; c0 += 16) {
    const float* Br = ckb + (size_t)(c0 + mrow) * cfg::DH;  // N-major source
    v8f acc = {};
    acc = __builtin_amdgcn_wmma_f32_16x16x32_bf16(false, a0, false, pack_bN(Br, hf),
                                                  (short)0, acc, false, false);
    acc = __builtin_amdgcn_wmma_f32_16x16x32_bf16(false, a1, false, pack_bN(Br + 32, hf),
                                                  (short)0, acc, false, false);
#pragma unroll
    for (int r = 0; r < 8; ++r)
      P[(size_t)(r + 8 * hf) * cfg::NCP + c0 + mrow] = acc[r] * scale;
  }
  __syncthreads();

  // phase 2: masked softmax + top-2 selection (lanes 0..15, one row each)
  if (lane < 16) {
    const int t = s0 + lane;
    float* row = P + (size_t)lane * cfg::NCP;
    int nvs = (t >= 3) ? (((t - 3) >> 1) + 1) : 0;   // max valid slot index
    if (nvs > cfg::NC) nvs = cfg::NC;
    float mx = row[0];
    for (int n = 1; n <= nvs; ++n) mx = fmaxf(mx, row[n]);
    float sum = 0.f;
    for (int n = 0; n < cfg::NCP; ++n) {
      float p = (n <= nvs) ? expf(row[n] - mx) : 0.f;
      row[n] = p; sum += p;
    }
    float inv = 1.f / sum;
    for (int n = 0; n < cfg::NCP; ++n) row[n] *= inv;
    // sel_scores via banded ov; imp[j] = row[j+1]
    float v1 = -1.f, v2 = -1.f; int i1 = 0, i2 = 0;
    for (int mm = 0; mm < cfg::S / cfg::SW; ++mm) {
      float sv = row[2 * mm + 1];
      if (mm > 0)                      sv += 0.5f * row[2 * mm];
      if (mm < cfg::S / cfg::SW - 1)   sv += 0.5f * row[2 * mm + 2];
      if (sv > v1)      { v2 = v1; i2 = i1; v1 = sv; i1 = mm; }
      else if (sv > v2) { v2 = sv; i2 = mm; }
    }
    size_t ti = (((size_t)b * cfg::H + hh) * cfg::S + t) * 2;
    topi[ti] = i1; topi[ti + 1] = i2;
  }
  __syncthreads();

  // phase 3: c_out = P @ CV (B operand contiguous via cvT)
  for (int dh0 = 0; dh0 < cfg::DH; dh0 += 16) {
    v8f acc = {};
    const float* Bc = cvTb + (size_t)(dh0 + mrow) * cfg::NCP;
    const float* Ap = P + (size_t)mrow * cfg::NCP;
    for (int k0 = 0; k0 < cfg::NCP; k0 += 32) {
      acc = __builtin_amdgcn_wmma_f32_16x16x32_bf16(
          false, pack_a(Ap + k0, hf), false, pack_bN(Bc + k0, hf),
          (short)0, acc, false, false);
    }
#pragma unroll
    for (int r = 0; r < 8; ++r) {
      int srow = s0 + r + 8 * hf;
      c_out[(((size_t)b * cfg::H + hh) * cfg::S + srow) * cfg::DH + dh0 + mrow] = acc[r];
    }
  }
}

// ---------------------------------------------------------------------------
// Gates: sigmoid(xn @ Wg + bg), M = 24 (too narrow for WMMA tiles)
// ---------------------------------------------------------------------------
__global__ void __launch_bounds__(256)
gates_kernel(const float* __restrict__ xn, const float* __restrict__ Wg,
             const float* __restrict__ bg, float* __restrict__ gates) {
  size_t idx = (size_t)blockIdx.x * blockDim.x + threadIdx.x;
  if (idx >= (size_t)cfg::B * cfg::S * cfg::G3H) return;
  int j = idx % cfg::G3H;
  size_t row = idx / cfg::G3H;
  const float* xr = xn + row * cfg::D;
  float acc = bg[j];
  for (int e = 0; e < cfg::D; ++e) acc += xr[e] * Wg[(size_t)e * cfg::G3H + j];
  gates[idx] = 1.f / (1.f + expf(-acc));
}

// ---------------------------------------------------------------------------
// Fine (top-2 blocks + current) + sliding-window attention + gated combine.
// One wave per (b,h,s) query row.
// ---------------------------------------------------------------------------
__global__ void __launch_bounds__(32)
fine_sw_kernel(const float* __restrict__ qr, const float* __restrict__ kr,
               const float* __restrict__ v, const float* __restrict__ c_out,
               const int* __restrict__ topi, const float* __restrict__ gates,
               float* __restrict__ att) {
  __shared__ float sim[16];
  const int wid = blockIdx.x;                  // B*H*S
  const int s  = wid % cfg::S;
  const int hh = (wid / cfg::S) % cfg::H;
  const int b  = wid / (cfg::S * cfg::H);
  const int lane = threadIdx.x;
  const float scale = 0.125f;
  constexpr int F = (cfg::NSEL + 1) * cfg::SW; // 12

  const float* qrow = qr + ((size_t)b * cfg::S + s) * cfg::D + (size_t)hh * cfg::DH;
  size_t ti = (((size_t)b * cfg::H + hh) * cfg::S + s) * 2;
  const int t0 = topi[ti], t1 = topi[ti + 1];
  const int cur = s / cfg::SW;
  int blocks[3] = { t0, t1, cur };
  bool ok[3] = { t0 != cur, t1 != cur, true };

  if (lane < F) {
    int slot = lane >> 2, i = lane & 3;
    int tok = blocks[slot] * cfg::SW + i;
    bool m = ok[slot] && (tok <= s);
    float d = kNEG;
    if (m) {
      const float* kro = kr + ((size_t)b * cfg::S + tok) * cfg::D + (size_t)hh * cfg::DH;
      float acc = 0.f;
      for (int e = 0; e < cfg::DH; ++e) acc += qrow[e] * kro[e];
      d = acc * scale;
    }
    sim[lane] = d;
  } else if (lane < F + cfg::WIN) {
    int w = lane - F;
    int widx = s - (cfg::WIN - 1) + w;
    bool m = widx >= 0;
    int tok = m ? widx : 0;
    const float* kro = kr + ((size_t)b * cfg::S + tok) * cfg::D + (size_t)hh * cfg::DH;
    float acc = 0.f;
    for (int e = 0; e < cfg::DH; ++e) acc += qrow[e] * kro[e];
    sim[lane] = m ? acc * scale : kNEG;
  }
  __syncthreads();

  // redundant per-lane softmax over tiny rows
  float p[F], pw[cfg::WIN];
  {
    float mx = kNEG;
    for (int f = 0; f < F; ++f) mx = fmaxf(mx, sim[f]);
    float ssum = 0.f;
    for (int f = 0; f < F; ++f) { p[f] = expf(sim[f] - mx); ssum += p[f]; }
    float inv = 1.f / ssum;
    for (int f = 0; f < F; ++f) p[f] *= inv;
    float mx2 = fmaxf(sim[F], sim[F + 1]);
    pw[0] = expf(sim[F] - mx2); pw[1] = expf(sim[F + 1] - mx2);
    float i2 = 1.f / (pw[0] + pw[1]);
    pw[0] *= i2; pw[1] *= i2;
  }

  const float* gp = gates + ((size_t)b * cfg::S + s) * cfg::G3H + (size_t)hh * 3;
  const float g0 = gp[0], g1 = gp[1], g2 = gp[2];
  const int w0 = (s - 1 >= 0) ? s - 1 : 0;

  for (int dh = lane; dh < cfg::DH; dh += 32) {
    float fo = 0.f;
    for (int f = 0; f < F; ++f) {
      int slot = f >> 2, i = f & 3;
      int tok = blocks[slot] * cfg::SW + i;
      fo += p[f] * v[((size_t)b * cfg::S + tok) * cfg::D + (size_t)hh * cfg::DH + dh];
    }
    float so = pw[0] * v[((size_t)b * cfg::S + w0) * cfg::D + (size_t)hh * cfg::DH + dh]
             + pw[1] * v[((size_t)b * cfg::S + s)  * cfg::D + (size_t)hh * cfg::DH + dh];
    float co = c_out[(((size_t)b * cfg::H + hh) * cfg::S + s) * cfg::DH + dh];
    att[((size_t)b * cfg::S + s) * cfg::D + (size_t)hh * cfg::DH + dh] =
        g0 * co + g1 * fo + g2 * so;
  }
}

__global__ void __launch_bounds__(256)
copy_f32(const float* __restrict__ src, float* __restrict__ dst, size_t n) {
  size_t i = (size_t)blockIdx.x * blockDim.x + threadIdx.x;
  if (i < n) dst[i] = src[i];
}

// ---------------------------------------------------------------------------
// Host orchestration
// ---------------------------------------------------------------------------
extern "C" void kernel_launch(void* const* d_in, const int* in_sizes, int n_in,
                              void* d_out, int out_size, void* d_ws, size_t ws_size,
                              hipStream_t stream) {
  using namespace cfg;
  (void)in_sizes; (void)n_in; (void)out_size; (void)ws_size;

  const float* src    = (const float*)d_in[0];
  const float* ln1_g  = (const float*)d_in[1];
  const float* ln1_b  = (const float*)d_in[2];
  const float* ln2_g  = (const float*)d_in[3];
  const float* ln2_b  = (const float*)d_in[4];
  const float* an_g   = (const float*)d_in[5];
  const float* Wq     = (const float*)d_in[6];
  const float* Wk     = (const float*)d_in[7];
  const float* Wv     = (const float*)d_in[8];
  const float* k_pos  = (const float*)d_in[9];
  const float* v_pos  = (const float*)d_in[10];
  const float* Wck    = (const float*)d_in[11];
  const float* bck    = (const float*)d_in[12];
  const float* Wcv    = (const float*)d_in[13];
  const float* bcv    = (const float*)d_in[14];
  const float* mem_k  = (const float*)d_in[15];
  const float* mem_v  = (const float*)d_in[16];
  const float* Wg     = (const float*)d_in[17];
  const float* bg     = (const float*)d_in[18];
  const float* Wo     = (const float*)d_in[19];
  const float* bo     = (const float*)d_in[20];
  const float* W1     = (const float*)d_in[21];
  const float* b1     = (const float*)d_in[22];
  const float* W2     = (const float*)d_in[23];
  const float* b2     = (const float*)d_in[24];

  // workspace layout (floats)
  float* wsf = (float*)d_ws;
  size_t off = 0;
  auto take = [&](size_t n) { float* p = wsf + off; off += n; return p; };
  float* x     = take((size_t)N * D);
  float* hbuf  = take((size_t)N * D);
  float* xn    = take((size_t)N * D);
  float* q     = take((size_t)N * D);
  float* k     = take((size_t)N * D);
  float* v     = take((size_t)N * D);
  float* qr    = take((size_t)N * D);
  float* kr    = take((size_t)N * D);
  float* c_out = take((size_t)N * D);
  float* att   = take((size_t)N * D);
  float* ck    = take((size_t)B * H * NCP * DH);
  float* cvT   = take((size_t)B * H * DH * NCP);
  float* ckc   = take((size_t)NR * DH);
  float* cvc   = take((size_t)NR * DH);
  float* gates = take((size_t)B * S * G3H);
  float* wt    = take((size_t)D * FF);          // transposed-weight staging
  int*   topi  = (int*)take((size_t)B * H * S * 2);
  // big scratch: comp rows during attention, FFN hidden afterwards
  float* scr   = take((size_t)N * FF);          // >= 2*NR*256
  float* kbr   = scr;
  float* vbr   = scr + (size_t)NR * (CW * DH);
  float* ff1   = scr;

  const size_t nelem = (size_t)N * D;
  copy_f32<<<dim3((nelem + 255) / 256), dim3(256), 0, stream>>>(src, x, nelem);

  auto transpose = [&](const float* W, int K_, int M_) {
    size_t tot = (size_t)K_ * M_;
    transpose_w<<<dim3((tot + 255) / 256), dim3(256), 0, stream>>>(W, wt, K_, M_);
  };

  for (int l = 0; l < L; ++l) {
    const float* ln1g = ln1_g + (size_t)l * D;
    const float* ln1b = ln1_b + (size_t)l * D;
    const float* ln2g = ln2_g + (size_t)l * D;
    const float* ln2b = ln2_b + (size_t)l * D;
    const float* ang  = an_g  + (size_t)l * D;
    const float* Wq_l = Wq + (size_t)l * D * D;
    const float* Wk_l = Wk + (size_t)l * D * D;
    const float* Wv_l = Wv + (size_t)l * D * D;
    const float* kp_l = k_pos + (size_t)l * H * CW * DH;
    const float* vp_l = v_pos + (size_t)l * H * CW * DH;
    const float* Wck_l = Wck + (size_t)l * (CW * DH) * DH;
    const float* bck_l = bck + (size_t)l * DH;
    const float* Wcv_l = Wcv + (size_t)l * (CW * DH) * DH;
    const float* bcv_l = bcv + (size_t)l * DH;
    const float* mk_l  = mem_k + (size_t)l * H * DH;
    const float* mv_l  = mem_v + (size_t)l * H * DH;
    const float* Wg_l  = Wg + (size_t)l * D * G3H;
    const float* bg_l  = bg + (size_t)l * G3H;
    const float* Wo_l  = Wo + (size_t)l * D * D;
    const float* bo_l  = bo + (size_t)l * D;
    const float* W1_l  = W1 + (size_t)l * D * FF;
    const float* b1_l  = b1 + (size_t)l * FF;
    const float* W2_l  = W2 + (size_t)l * FF * D;
    const float* b2_l  = b2 + (size_t)l * D;

    // norms
    layernorm_kernel<<<dim3(N), dim3(256), 0, stream>>>(x, ln1g, ln1b, hbuf);
    rmsnorm_kernel<<<dim3(N), dim3(256), 0, stream>>>(hbuf, ang, xn);

    // QKV projections (WMMA, 2x2 blocked, weights transposed first)
    transpose(Wq_l, D, D);
    gemm_bf16_wmma<0, 2, 2><<<dim3(D / 32, N / 32), dim3(32), 0, stream>>>(xn, wt, nullptr, nullptr, q, D, D);
    transpose(Wk_l, D, D);
    gemm_bf16_wmma<0, 2, 2><<<dim3(D / 32, N / 32), dim3(32), 0, stream>>>(xn, wt, nullptr, nullptr, k, D, D);
    transpose(Wv_l, D, D);
    gemm_bf16_wmma<0, 2, 2><<<dim3(D / 32, N / 32), dim3(32), 0, stream>>>(xn, wt, nullptr, nullptr, v, D, D);

    // compression rows + compression GEMMs (WMMA) + assemble (mem slot, cv^T)
    {
      size_t tot = (size_t)B * H * NC * (CW * DH);
      build_comp_rows<<<dim3((tot + 255) / 256), dim3(256), 0, stream>>>(k, v, kp_l, vp_l, kbr, vbr);
    }
    transpose(Wck_l, CW * DH, DH);
    gemm_bf16_wmma<1, 1, 2><<<dim3(DH / 32, NR / 16), dim3(32), 0, stream>>>(kbr, wt, bck_l, nullptr, ckc, CW * DH, DH);
    transpose(Wcv_l, CW * DH, DH);
    gemm_bf16_wmma<1, 1, 2><<<dim3(DH / 32, NR / 16), dim3(32), 0, stream>>>(vbr, wt, bcv_l, nullptr, cvc, CW * DH, DH);
    {
      size_t tot = (size_t)B * H * NCP * DH;
      assemble_comp<<<dim3((tot + 255) / 256), dim3(256), 0, stream>>>(ckc, cvc, mk_l, mv_l, ck, cvT);
    }

    // rope
    {
      size_t tot = (size_t)B * S * H * 32;
      rope_kernel<<<dim3((tot + 255) / 256), dim3(256), 0, stream>>>(q, k, qr, kr);
    }

    // compressed attention (WMMA, 64KB dynamic LDS per workgroup)
    cattn_kernel<<<dim3(B * H * (S / 16)), dim3(32), 16 * NCP * sizeof(float), stream>>>(q, ck, cvT, c_out, topi);

    // gates, fine + sliding + combine
    {
      size_t tot = (size_t)B * S * G3H;
      gates_kernel<<<dim3((tot + 255) / 256), dim3(256), 0, stream>>>(xn, Wg_l, bg_l, gates);
    }
    fine_sw_kernel<<<dim3(B * H * S), dim3(32), 0, stream>>>(qr, kr, v, c_out, topi, gates, att);

    // output projection + residual (WMMA)
    transpose(Wo_l, D, D);
    gemm_bf16_wmma<3, 2, 2><<<dim3(D / 32, N / 32), dim3(32), 0, stream>>>(att, wt, bo_l, x, x, D, D);

    // FFN (WMMA): GELU(h @ W1 + b1) @ W2 + b2 + residual
    layernorm_kernel<<<dim3(N), dim3(256), 0, stream>>>(x, ln2g, ln2b, hbuf);
    transpose(W1_l, D, FF);
    gemm_bf16_wmma<2, 2, 2><<<dim3(FF / 32, N / 32), dim3(32), 0, stream>>>(hbuf, wt, b1_l, nullptr, ff1, D, FF);
    transpose(W2_l, FF, D);
    gemm_bf16_wmma<3, 2, 2><<<dim3(D / 32, N / 32), dim3(32), 0, stream>>>(ff1, wt, b2_l, x, x, FF, D);
  }

  copy_f32<<<dim3((nelem + 255) / 256), dim3(256), 0, stream>>>(x, (float*)d_out, nelem);
}
